// SpGAT_45767171506713
// MI455X (gfx1250) — compile-verified, matrix operand order checked by reading, not run on hardware
//
#include <hip/hip_runtime.h>

typedef __attribute__((ext_vector_type(16))) _Float16 v16h;
typedef __attribute__((ext_vector_type(8)))  float    v8f;

static constexpr int N_NODES = 50000;
static constexpr int NFEAT   = 64;
static constexpr int NHID    = 64;
static constexpr int REL_DIM = 64;
static constexpr int NHEADS  = 2;
static constexpr int N_REL   = 200;
static constexpr int N_EDGES = 200000;
static constexpr int N_NHOP  = 40000;
static constexpr int E_TOT   = N_EDGES + N_NHOP;   // 240000
static constexpr int OUTF    = NHEADS * NHID;      // 128
static constexpr float ALPHA = 0.2f;

__device__ __forceinline__ void fatomic_add(float* p, float v) {
  __hip_atomic_fetch_add(p, v, __ATOMIC_RELAXED, __HIP_MEMORY_SCOPE_AGENT);
}

// per-lane gather straight into LDS via the gfx1250 async copy path
__device__ __forceinline__ void async_gather_b128(unsigned lds_byte_off, const void* gptr) {
  asm volatile("global_load_async_to_lds_b128 %0, %1, off"
               :: "v"(lds_byte_off), "v"((unsigned long long)(uintptr_t)gptr)
               : "memory");
}
__device__ __forceinline__ void wait_async0() {
  asm volatile("s_wait_asynccnt 0x0" ::: "memory");
}

// ---------------- utility kernels ----------------

__global__ void k_zero(float* __restrict__ p, long n) {
  long i = (long)blockIdx.x * blockDim.x + threadIdx.x;
  long stride = (long)gridDim.x * blockDim.x;
  for (; i < n; i += stride) p[i] = 0.f;
}

__global__ void k_f32_to_f16(const float* __restrict__ src, _Float16* __restrict__ dst, long n) {
  long i = (long)blockIdx.x * blockDim.x + threadIdx.x;
  if (i < n) dst[i] = (_Float16)src[i];
}

// eemb1h[e][k], k<64: edges then nhop (rel[t0]+rel[t1])
__global__ void k_eemb1(const float* __restrict__ edge_embed, const float* __restrict__ rel,
                        const int* __restrict__ etn, _Float16* __restrict__ dst) {
  long i = (long)blockIdx.x * blockDim.x + threadIdx.x;
  if (i >= (long)E_TOT * REL_DIM) return;
  int e = (int)(i >> 6), k = (int)(i & 63);
  float v;
  if (e < N_EDGES) v = edge_embed[i];
  else {
    int e2 = e - N_EDGES;
    v = rel[etn[e2 * 2] * REL_DIM + k] + rel[etn[e2 * 2 + 1] * REL_DIM + k];
  }
  dst[i] = (_Float16)v;
}

// out_relation_1 = rel(200x64) @ W(64x128); write to ws + d_out tail
__global__ void k_outrel(const float* __restrict__ rel, const float* __restrict__ W,
                         float* __restrict__ outrel, float* __restrict__ outTail) {
  int i = blockIdx.x * blockDim.x + threadIdx.x;
  if (i >= N_REL * OUTF) return;
  int r = i >> 7, j = i & 127;
  float s = 0.f;
  for (int k = 0; k < REL_DIM; ++k) s = fmaf(rel[r * REL_DIM + k], W[k * OUTF + j], s);
  outrel[i] = s;
  outTail[i] = s;
}

// eemb2h[e][j], j<128: outrel gathered by edge type
__global__ void k_eemb2(const float* __restrict__ outrel, const int* __restrict__ et,
                        const int* __restrict__ etn, _Float16* __restrict__ dst) {
  long i = (long)blockIdx.x * blockDim.x + threadIdx.x;
  if (i >= (long)E_TOT * OUTF) return;
  int e = (int)(i >> 7), j = (int)(i & 127);
  float v;
  if (e < N_EDGES) v = outrel[(size_t)et[e] * OUTF + j];
  else {
    int e2 = e - N_EDGES;
    v = outrel[(size_t)etn[e2 * 2] * OUTF + j] + outrel[(size_t)etn[e2 * 2 + 1] * OUTF + j];
  }
  dst[i] = (_Float16)v;
}

// concat edge lists into flat src/dst arrays
__global__ void k_eall(const int* __restrict__ el, const int* __restrict__ eln,
                       int* __restrict__ e0, int* __restrict__ e1) {
  int i = blockIdx.x * blockDim.x + threadIdx.x;
  if (i >= E_TOT) return;
  if (i < N_EDGES) { e0[i] = el[i]; e1[i] = el[N_EDGES + i]; }
  else             { e0[i] = eln[i - N_EDGES]; e1[i] = eln[N_NHOP + (i - N_EDGES)]; }
}

// pack attention matrix a (NROWS x K, row-major) into WMMA B-fragment order:
// frag[t][c][lane][i], col = lane&15 -> N index t*16+col,
// k = c*32 + (lane>=16 ? 8 : 0) + (i<8 ? i : i+8)   (16-bit pair interleave)
__global__ void k_bfrag(const float* __restrict__ a, _Float16* __restrict__ frag,
                        int CH, int K, int total) {
  int idx = blockIdx.x * blockDim.x + threadIdx.x;
  if (idx >= total) return;
  int i = idx & 15;
  int lane = (idx >> 4) & 31;
  int rest = idx >> 9;
  int c = rest % CH;
  int t = rest / CH;
  int row = t * 16 + (lane & 15);
  int k = c * 32 + ((lane >> 4) << 3) + (i < 8 ? i : i + 8);
  frag[idx] = (_Float16)a[(size_t)row * K + k];
}

__global__ void k_WeT(const float* __restrict__ We, float* __restrict__ WeT) {
  int i = blockIdx.x * blockDim.x + threadIdx.x;      // over 64*128
  if (i >= NHID * OUTF) return;
  int j = i >> 7, k = i & 127;
  WeT[i] = We[k * NHID + j];
}

// ---------------- WMMA edge kernel ----------------
// One wave = 16 edges. h = [x[key] | x[oth] | eemb] (f16), K = 2*XDIM+EDIM.
// Staging: global_load_async_to_lds_b128 per-lane gather (ASYNCcnt).
// D = h @ a^T via v_wmma_f32_16x16x32_f16, then attention exp(-leaky_relu(D@a2))
// and atomic scatter into xacc[node][128] and rowsum[head][node].
template <int XDIM, int EDIM, int NH, int NOUT, int WAVES>
__launch_bounds__(WAVES * 32)
__global__ void k_edge_gat(const _Float16* __restrict__ xh,
                           const _Float16* __restrict__ eembh,
                           const _Float16* __restrict__ bfrag,
                           const float* __restrict__ a2flat,
                           const int* __restrict__ key,
                           const int* __restrict__ oth,
                           float* __restrict__ xacc,
                           float* __restrict__ rowsum) {
  constexpr int K    = 2 * XDIM + EDIM;
  constexpr int CH   = K / 32;
  constexpr int NTT  = (NH * NOUT) / 16;   // 8 for both layers
  constexpr int TPH  = NTT / NH;
  constexpr int KSTR = K + 8;              // f16 stride (16B-aligned rows)
  constexpr int KV   = K / 8;              // uint4 per edge row
  constexpr int XV   = XDIM / 8;

  __shared__ alignas(16) _Float16 sh[WAVES * 16 * KSTR];
  __shared__ int snode[WAVES * 16];

  const int lane  = threadIdx.x & 31;
  const int wave  = threadIdx.x >> 5;
  const int tile  = blockIdx.x * WAVES + wave;
  const int ebase = tile * 16;

  _Float16* ht = sh + wave * 16 * KSTR;
  int* nd = snode + wave * 16;

  if (lane < 16) nd[lane] = key[ebase + lane];

  // stage the 16 x K f16 edge-feature tile directly into LDS (async copy path)
  const unsigned ldsbase = (unsigned)(uintptr_t)ht;   // low 32 bits = LDS byte offset
  for (int f = lane; f < 16 * KV; f += 32) {
    const int er = f / KV;
    const int q  = f - er * KV;
    const int e  = ebase + er;
    const _Float16* gp;
    if (q < XV)            gp = xh + (size_t)key[e] * XDIM + (size_t)q * 8;
    else if (q < 2 * XV)   gp = xh + (size_t)oth[e] * XDIM + (size_t)(q - XV) * 8;
    else                   gp = eembh + (size_t)e * EDIM + (size_t)(q - 2 * XV) * 8;
    async_gather_b128(ldsbase + (unsigned)((er * KSTR + q * 8) * 2), gp);
  }
  wait_async0();   // ht tile resident; each wave reads only its own slice

  const int col   = lane & 15;
  const int half  = lane >> 4;
  const int rbase = half << 3;   // rows rbase..rbase+7 held by this lane

  v8f acc[NTT] = {};
  union AB { uint4 u[2]; v16h h; };

  const uint4* bf4 = reinterpret_cast<const uint4*>(bfrag);
#pragma unroll 1
  for (int c = 0; c < CH; ++c) {
    // A fragment (16x32 f16): lane holds row=col, K = c*32 + half*8 + {0..7, 16..23}
    const _Float16* ap = ht + col * KSTR + c * 32 + (half << 3);
    AB a;
    a.u[0] = *reinterpret_cast<const uint4*>(ap);
    a.u[1] = *reinterpret_cast<const uint4*>(ap + 16);
    const uint4* bp = bf4 + ((size_t)c * 32 + lane) * 2;
#pragma unroll
    for (int t = 0; t < NTT; ++t) {
      AB b;
      const uint4* bt = bp + (size_t)t * CH * 32 * 2;
      b.u[0] = bt[0];
      b.u[1] = bt[1];
      acc[t] = __builtin_amdgcn_wmma_f32_16x16x32_f16(false, a.h, false, b.h,
                                                      (short)0, acc[t], false, false);
    }
  }

  // attention: s[row] = sum_n D[row][n]*a2[n]; e = exp(-leaky_relu(s))
  float wt[NTT];
#pragma unroll
  for (int t = 0; t < NTT; ++t) wt[t] = a2flat[t * 16 + col];

  float eatt[NH][8];
#pragma unroll
  for (int h = 0; h < NH; ++h) {
#pragma unroll
    for (int r = 0; r < 8; ++r) {
      float p = 0.f;
#pragma unroll
      for (int tt = 0; tt < TPH; ++tt) p += acc[h * TPH + tt][r] * wt[h * TPH + tt];
      p += __shfl_xor(p, 1);
      p += __shfl_xor(p, 2);
      p += __shfl_xor(p, 4);
      p += __shfl_xor(p, 8);
      const float lr = p > 0.f ? p : ALPHA * p;
      eatt[h][r] = __expf(-lr);
    }
  }

  if (col < 8) {
    const int row  = rbase + col;
    const int node = nd[row];
#pragma unroll
    for (int h = 0; h < NH; ++h)
      fatomic_add(&rowsum[(size_t)h * N_NODES + node], eatt[h][col]);
  }

#pragma unroll
  for (int t = 0; t < NTT; ++t) {
    const int h = t / TPH;
#pragma unroll
    for (int r = 0; r < 8; ++r) {
      const int node = nd[rbase + r];
      fatomic_add(&xacc[(size_t)node * (NTT * 16) + t * 16 + col], acc[t][r] * eatt[h][r]);
    }
  }
}

// normalize + elu; optional f16 mirror for next layer's gathers
__global__ void k_norm(const float* __restrict__ xacc, const float* __restrict__ rowsum,
                       int noutShift, float* __restrict__ xs32, _Float16* __restrict__ xs16) {
  long i = (long)blockIdx.x * blockDim.x + threadIdx.x;
  if (i >= (long)N_NODES * OUTF) return;
  int node = (int)(i >> 7);
  int c = (int)(i & 127);
  int h = c >> noutShift;
  float rs = rowsum[(size_t)h * N_NODES + node];
  rs = (rs == 0.f) ? 1e-12f : rs;
  float v = xacc[i] / rs;
  v = v > 0.f ? v : (__expf(v) - 1.f);
  xs32[i] = v;
  if (xs16) xs16[i] = (_Float16)v;
}

// per-node: sum_j tanh(x@We[:,j]+be[j]) * qe[j], reduced into wslot
__global__ void k_attn_pool(const float* __restrict__ xf, const float* __restrict__ WeT,
                            const float* __restrict__ be, const float* __restrict__ qe,
                            float* __restrict__ wslot) {
  int node = blockIdx.x * blockDim.x + threadIdx.x;
  if (node >= N_NODES) return;
  float xr[OUTF];
  const float4* xp = reinterpret_cast<const float4*>(xf + (size_t)node * OUTF);
#pragma unroll
  for (int k = 0; k < OUTF / 4; ++k) {
    float4 v = xp[k];
    xr[4 * k] = v.x; xr[4 * k + 1] = v.y; xr[4 * k + 2] = v.z; xr[4 * k + 3] = v.w;
  }
  float accv = 0.f;
  for (int j = 0; j < NHID; ++j) {
    float t = be[j];
    const float* wr = WeT + (size_t)j * OUTF;
#pragma unroll 8
    for (int k = 0; k < OUTF; ++k) t = fmaf(xr[k], wr[k], t);
    accv += tanhf(t) * qe[j];
  }
  fatomic_add(wslot, accv);
}

__global__ void k_combine(const float* __restrict__ xf0, const float* __restrict__ xf1,
                          const float* __restrict__ wsum, float* __restrict__ out) {
  long i = (long)blockIdx.x * blockDim.x + threadIdx.x;
  if (i >= (long)N_NODES * OUTF) return;
  float m0 = wsum[0] / (float)N_NODES, m1 = wsum[1] / (float)N_NODES;
  float mx = fmaxf(m0, m1);
  float b0 = __expf(m0 - mx), b1 = __expf(m1 - mx);
  float inv = 1.f / (b0 + b1);
  out[i] = (b0 * xf0[i] + b1 * xf1[i]) * inv;
}

// ---------------- host ----------------

extern "C" void kernel_launch(void* const* d_in, const int* in_sizes, int n_in,
                              void* d_out, int out_size, void* d_ws, size_t ws_size,
                              hipStream_t stream) {
  (void)in_sizes; (void)n_in; (void)out_size; (void)ws_size;
  const float* entity = (const float*)d_in[0];
  const float* rel    = (const float*)d_in[1];
  const int*   el     = (const int*)d_in[2];
  const int*   et     = (const int*)d_in[3];
  const float* eemb   = (const float*)d_in[4];
  const int*   eln    = (const int*)d_in[5];
  const int*   etn    = (const int*)d_in[6];
  const float* a_in   = (const float*)d_in[7];
  const float* a2_in  = (const float*)d_in[8];
  const float* a_out  = (const float*)d_in[9];
  const float* a2_out = (const float*)d_in[10];
  const float* a_o    = (const float*)d_in[11];
  const float* a2_o   = (const float*)d_in[12];
  const float* W      = (const float*)d_in[13];
  const float* We     = (const float*)d_in[14];
  const float* be     = (const float*)d_in[15];
  const float* qe     = (const float*)d_in[16];
  float* outp = (float*)d_out;

  char* p = (char*)d_ws;
  auto alloc = [&](size_t bytes) { char* r = p; p += (bytes + 255) & ~(size_t)255; return r; };

  _Float16* xh     = (_Float16*)alloc((size_t)N_NODES * NFEAT * 2);
  _Float16* eemb1h = (_Float16*)alloc((size_t)E_TOT * REL_DIM * 2);
  _Float16* eemb2h = (_Float16*)alloc((size_t)E_TOT * OUTF * 2);
  _Float16* bfin   = (_Float16*)alloc((size_t)8 * 6 * 32 * 16 * 2);
  _Float16* bfout  = (_Float16*)alloc((size_t)8 * 6 * 32 * 16 * 2);
  _Float16* bfo    = (_Float16*)alloc((size_t)8 * 12 * 32 * 16 * 2);
  int* e0          = (int*)alloc((size_t)E_TOT * 4);
  int* e1          = (int*)alloc((size_t)E_TOT * 4);
  float* xacc      = (float*)alloc((size_t)2 * N_NODES * OUTF * 4);
  float* rowsum    = (float*)alloc((size_t)2 * NHEADS * N_NODES * 4);
  float* xs32      = (float*)alloc((size_t)2 * N_NODES * OUTF * 4);
  _Float16* xs16   = (_Float16*)alloc((size_t)2 * N_NODES * OUTF * 2);
  float* xf        = (float*)alloc((size_t)2 * N_NODES * OUTF * 4);
  float* outrel    = (float*)alloc((size_t)N_REL * OUTF * 4);
  float* WeT       = (float*)alloc((size_t)NHID * OUTF * 4);
  float* wsum      = (float*)alloc(2 * 4);

  const int B = 256;
  auto g = [&](long n) { return (int)((n + B - 1) / B); };
  const size_t DS = (size_t)N_NODES * OUTF;   // per-direction stride

  // prep
  k_f32_to_f16<<<g((long)N_NODES * NFEAT), B, 0, stream>>>(entity, xh, (long)N_NODES * NFEAT);
  k_eemb1<<<g((long)E_TOT * REL_DIM), B, 0, stream>>>(eemb, rel, etn, eemb1h);
  k_bfrag<<<g(8 * 6 * 32 * 16), B, 0, stream>>>(a_in, bfin, 6, 192, 8 * 6 * 32 * 16);
  k_bfrag<<<g(8 * 6 * 32 * 16), B, 0, stream>>>(a_out, bfout, 6, 192, 8 * 6 * 32 * 16);
  k_bfrag<<<g(8 * 12 * 32 * 16), B, 0, stream>>>(a_o, bfo, 12, 384, 8 * 12 * 32 * 16);
  k_outrel<<<g(N_REL * OUTF), B, 0, stream>>>(rel, W, outrel, outp + DS);
  k_eemb2<<<g((long)E_TOT * OUTF), B, 0, stream>>>(outrel, et, etn, eemb2h);
  k_eall<<<g(E_TOT), B, 0, stream>>>(el, eln, e0, e1);
  k_WeT<<<g(NHID * OUTF), B, 0, stream>>>(We, WeT);

  // layer 1 (2 heads, NOUT=64, K=192)
  k_zero<<<g((long)2 * DS), B, 0, stream>>>(xacc, (long)2 * DS);
  k_zero<<<g((long)2 * NHEADS * N_NODES), B, 0, stream>>>(rowsum, (long)2 * NHEADS * N_NODES);
  dim3 gl1(E_TOT / (16 * 8));
  k_edge_gat<64, 64, 2, 64, 8><<<gl1, 256, 0, stream>>>(xh, eemb1h, bfin, a2_in, e0, e1,
                                                        xacc, rowsum);
  k_edge_gat<64, 64, 2, 64, 8><<<gl1, 256, 0, stream>>>(xh, eemb1h, bfout, a2_out, e1, e0,
                                                        xacc + DS, rowsum + (size_t)NHEADS * N_NODES);
  k_norm<<<g((long)DS), B, 0, stream>>>(xacc, rowsum, 6, xs32, xs16);
  k_norm<<<g((long)DS), B, 0, stream>>>(xacc + DS, rowsum + (size_t)NHEADS * N_NODES, 6,
                                        xs32 + DS, xs16 + DS);

  // layer 2 (1 "head", NOUT=128, K=384) — reuse xacc/rowsum
  k_zero<<<g((long)2 * DS), B, 0, stream>>>(xacc, (long)2 * DS);
  k_zero<<<g((long)2 * NHEADS * N_NODES), B, 0, stream>>>(rowsum, (long)2 * NHEADS * N_NODES);
  dim3 gl2(E_TOT / (16 * 4));
  k_edge_gat<128, 128, 1, 128, 4><<<gl2, 128, 0, stream>>>(xs16, eemb2h, bfo, a2_o, e0, e1,
                                                           xacc, rowsum);
  k_edge_gat<128, 128, 1, 128, 4><<<gl2, 128, 0, stream>>>(xs16 + DS, eemb2h, bfo, a2_o, e1, e0,
                                                           xacc + DS, rowsum + (size_t)NHEADS * N_NODES);
  k_norm<<<g((long)DS), B, 0, stream>>>(xacc, rowsum, 7, xf, (_Float16*)nullptr);
  k_norm<<<g((long)DS), B, 0, stream>>>(xacc + DS, rowsum + (size_t)NHEADS * N_NODES, 7,
                                        xf + DS, (_Float16*)nullptr);

  // pooling + combine
  k_zero<<<1, 32, 0, stream>>>(wsum, 2);
  k_attn_pool<<<g(N_NODES), B, 0, stream>>>(xf, WeT, be, qe, &wsum[0]);
  k_attn_pool<<<g(N_NODES), B, 0, stream>>>(xf + DS, WeT, be, qe, &wsum[1]);
  k_combine<<<g((long)DS), B, 0, stream>>>(xf, xf + DS, wsum, outp);
}
